// ComponentNN_88665304858957
// MI455X (gfx1250) — compile-verified, matrix operand order checked by reading, not run on hardware
//
#include <hip/hip_runtime.h>

typedef _Float16 v16h __attribute__((ext_vector_type(16)));
typedef float    v8f  __attribute__((ext_vector_type(8)));
typedef _Float16 h2   __attribute__((ext_vector_type(2)));
typedef unsigned int u32x4 __attribute__((ext_vector_type(4)));

union Frag {
    v16h  f;
    h2    p[8];
    u32x4 q[2];
};

__device__ __forceinline__ void load_frag(Frag& fr, const _Float16* p) {
    fr.q[0] = *(const u32x4*)(p);
    fr.q[1] = *(const u32x4*)(p + 8);
}

// ---------------- prep kernels ----------------

__global__ void k_cvt_f16(const float* __restrict__ in, _Float16* __restrict__ out, long n) {
    long i = (long)blockIdx.x * blockDim.x + threadIdx.x;
    if (i < n) out[i] = (_Float16)in[i];
}

// W3 (E,363,512) f32 -> (E,384,512) f16, zero-padded rows 363..383
__global__ void k_pad_w3(const float* __restrict__ W3, _Float16* __restrict__ out, long n) {
    long idx = (long)blockIdx.x * blockDim.x + threadIdx.x;
    if (idx >= n) return;
    int i = (int)(idx % 512);
    long t = idx / 512;
    int o = (int)(t % 384);
    int e = (int)(t / 384);
    out[idx] = (o < 363) ? (_Float16)W3[((long)e * 363 + o) * 512 + i] : (_Float16)0.0f;
}

// blend (E,B) f32 -> blendT (B,32) f16, cols 8..31 zero  (A-matrix for bias WMMA)
__global__ void k_blendT(const float* __restrict__ blend, _Float16* __restrict__ out, int B, long n) {
    long idx = (long)blockIdx.x * blockDim.x + threadIdx.x;
    if (idx >= n) return;
    int k = (int)(idx & 31);
    int b = (int)(idx >> 5);
    out[idx] = (k < 8) ? (_Float16)blend[(long)k * B + b] : (_Float16)0.0f;
}

// bias (E,outValid) f32 -> biasB (outPad,32) f16: biasB[o*32+k] = bias[k*outValid+o]
__global__ void k_biasB(const float* __restrict__ bias, _Float16* __restrict__ out,
                        int outValid, int outPad, long n) {
    long idx = (long)blockIdx.x * blockDim.x + threadIdx.x;
    if (idx >= n) return;
    int k = (int)(idx & 31);
    int o = (int)(idx >> 5);
    out[idx] = (k < 8 && o < outValid) ? (_Float16)bias[(long)k * outValid + o] : (_Float16)0.0f;
}

// ---------------- blended GEMM ----------------
// C(B x NPAD) = sum_e [diag(blend[e,:]) * A16] @ W_e^T + blend^T @ biasStack
// A16: (B,K) f16 row-major.  W16: (E,NPAD,K) f16 row-major (= W_e in (out,in) order).
// K, NPAD compile-time so the expert stride e*NPAD*K becomes an imm24 load offset.
// MODE 0: ELU -> f16 store (stride outStride).  MODE 1: raw f32 store.
template <int MODE, int K, int NPAD>
__global__ __launch_bounds__(128)
void gemm_blend(const _Float16* __restrict__ A16,
                const _Float16* __restrict__ W16,
                const _Float16* __restrict__ blendT,
                const _Float16* __restrict__ biasB,
                const float* __restrict__ blend, int Bsz,
                void* __restrict__ outp, int outStride)
{
    const int lane = threadIdx.x & 31;
    const int wave = threadIdx.x >> 5;       // 0..3
    const int wm = wave >> 1, wn = wave & 1; // 2x2 wave grid
    const int r  = lane & 15;
    const int kh = lane & 16;                // 0 or 16 halves (K-half select)
    const int m0 = blockIdx.x * 128 + wm * 64;
    const int n0 = blockIdx.y * 128 + wn * 64;

    // packed per-row blend scalars for v_pk_mul_f16 scaling
    h2 blpk[4][8];
#pragma unroll
    for (int i = 0; i < 4; ++i) {
        int row = m0 + i * 16 + r;
#pragma unroll
        for (int e = 0; e < 8; ++e) {
            _Float16 h = (_Float16)blend[(long)e * Bsz + row];
            h2 v = {h, h};
            blpk[i][e] = v;
        }
    }

    v8f C[4][4];
    // accumulator init = blended bias via one K=32 WMMA per tile
    {
        Frag ab[4], bb[4];
#pragma unroll
        for (int i = 0; i < 4; ++i)
            load_frag(ab[i], blendT + (size_t)(m0 + i * 16 + r) * 32 + kh);
#pragma unroll
        for (int j = 0; j < 4; ++j)
            load_frag(bb[j], biasB + (size_t)(n0 + j * 16 + r) * 32 + kh);
#pragma unroll
        for (int i = 0; i < 4; ++i)
#pragma unroll
            for (int j = 0; j < 4; ++j) {
                v8f z = {};
                C[i][j] = __builtin_amdgcn_wmma_f32_16x16x32_f16(
                    false, ab[i].f, false, bb[j].f, (short)0, z, false, false);
            }
    }

    // per-lane row base pointers; bumped by +32 halves per k-step so all loads
    // inside the body use immediate offsets (incl. constant e*NPAD*K expert stride)
    const _Float16* ap[4];
#pragma unroll
    for (int i = 0; i < 4; ++i)
        ap[i] = A16 + (size_t)(m0 + i * 16 + r) * K + kh;
    const _Float16* wp[4];
#pragma unroll
    for (int j = 0; j < 4; ++j)
        wp[j] = W16 + (size_t)(n0 + j * 16 + r) * K + kh;

#pragma unroll 1
    for (int kb = 0; kb < K; kb += 32) {
        Frag A[4];
#pragma unroll
        for (int i = 0; i < 4; ++i) load_frag(A[i], ap[i]);
#pragma unroll
        for (int e = 0; e < 8; ++e) {
            Frag Bf[4];
#pragma unroll
            for (int j = 0; j < 4; ++j)
                load_frag(Bf[j], wp[j] + e * (NPAD * K));   // constant offset
#pragma unroll
            for (int i = 0; i < 4; ++i) {
                Frag As;
#pragma unroll
                for (int t = 0; t < 8; ++t) As.p[t] = A[i].p[t] * blpk[i][e];
#pragma unroll
                for (int j = 0; j < 4; ++j)
                    C[i][j] = __builtin_amdgcn_wmma_f32_16x16x32_f16(
                        false, As.f, false, Bf[j].f, (short)0, C[i][j], false, false);
            }
        }
#pragma unroll
        for (int i = 0; i < 4; ++i) ap[i] += 32;
#pragma unroll
        for (int j = 0; j < 4; ++j) wp[j] += 32;
    }

    // epilogue: C layout — VGPR t: row = t + (lane>=16 ? 8:0); col = lane&15
    const int rbase = (lane >> 4) << 3;
#pragma unroll
    for (int i = 0; i < 4; ++i)
#pragma unroll
        for (int j = 0; j < 4; ++j) {
            int colg = n0 + j * 16 + r;
#pragma unroll
            for (int t = 0; t < 8; ++t) {
                int rowg = m0 + i * 16 + rbase + t;
                float v = C[i][j][t];
                if (MODE == 0) {
                    float ev = v > 0.0f ? v : (__expf(v) - 1.0f);
                    ((_Float16*)outp)[(size_t)rowg * outStride + colg] = (_Float16)ev;
                } else {
                    ((float*)outp)[(size_t)rowg * outStride + colg] = v;
                }
            }
        }
}

// ---------------- softmax: one wave32 per row ----------------
__global__ __launch_bounds__(256)
void softmax_rows(const float* __restrict__ logits, float* __restrict__ out,
                  int rows, int cols, int stride)
{
    int row  = (int)((blockIdx.x * (long)blockDim.x + threadIdx.x) >> 5);
    int lane = threadIdx.x & 31;
    if (row >= rows) return;
    const float* rp = logits + (size_t)row * stride;
    float mx = -3.4e38f;
    for (int c = lane; c < cols; c += 32) mx = fmaxf(mx, rp[c]);
    for (int o = 16; o; o >>= 1) mx = fmaxf(mx, __shfl_xor(mx, o, 32));
    float sum = 0.0f;
    for (int c = lane; c < cols; c += 32) sum += __expf(rp[c] - mx);
    for (int o = 16; o; o >>= 1) sum += __shfl_xor(sum, o, 32);
    float inv = 1.0f / sum;
    float* op = out + (size_t)row * cols;
    for (int c = lane; c < cols; c += 32) op[c] = __expf(rp[c] - mx) * inv;
}

// ---------------- launch ----------------
extern "C" void kernel_launch(void* const* d_in, const int* in_sizes, int n_in,
                              void* d_out, int out_size, void* d_ws, size_t ws_size,
                              hipStream_t stream) {
    const float* x     = (const float*)d_in[0];
    const float* blend = (const float*)d_in[1];
    const float* W1    = (const float*)d_in[2];
    const float* b1    = (const float*)d_in[3];
    const float* W2    = (const float*)d_in[4];
    const float* b2    = (const float*)d_in[5];
    const float* W3    = (const float*)d_in[6];
    const float* b3    = (const float*)d_in[7];
    float* out = (float*)d_out;

    const int B = 8192, E = 8;
    const int D0 = 480, D1 = 512, D2 = 512, D3 = 363, D3P = 384;

    char* ws = (char*)d_ws;
    size_t off = 0;
    auto wsAlloc = [&](size_t bytes) -> void* {
        void* p = ws + off;
        off += (bytes + 255) & ~(size_t)255;
        return p;
    };
    _Float16* x16    = (_Float16*)wsAlloc((size_t)B * D0 * 2);
    _Float16* h1     = (_Float16*)wsAlloc((size_t)B * D1 * 2);
    _Float16* h2b    = (_Float16*)wsAlloc((size_t)B * D2 * 2);
    float*    logits = (float*)   wsAlloc((size_t)B * D3P * 4);
    _Float16* W1h    = (_Float16*)wsAlloc((size_t)E * D1 * D0 * 2);
    _Float16* W2h    = (_Float16*)wsAlloc((size_t)E * D2 * D1 * 2);
    _Float16* W3h    = (_Float16*)wsAlloc((size_t)E * D3P * D2 * 2);
    _Float16* blT    = (_Float16*)wsAlloc((size_t)B * 32 * 2);
    _Float16* b1B    = (_Float16*)wsAlloc((size_t)D1 * 32 * 2);
    _Float16* b2B    = (_Float16*)wsAlloc((size_t)D2 * 32 * 2);
    _Float16* b3B    = (_Float16*)wsAlloc((size_t)D3P * 32 * 2);
    (void)ws_size; (void)n_in; (void)in_sizes; (void)out_size;

    auto gsz = [](long n) { return (unsigned)((n + 255) / 256); };

    // f32 -> f16 conversions (W1/W2 linear layouts are identical to desired B layout)
    { long n = (long)B * D0;           k_cvt_f16<<<gsz(n), 256, 0, stream>>>(x,  x16, n); }
    { long n = (long)E * D1 * D0;      k_cvt_f16<<<gsz(n), 256, 0, stream>>>(W1, W1h, n); }
    { long n = (long)E * D2 * D1;      k_cvt_f16<<<gsz(n), 256, 0, stream>>>(W2, W2h, n); }
    { long n = (long)E * D3P * D2;     k_pad_w3 <<<gsz(n), 256, 0, stream>>>(W3, W3h, n); }
    { long n = (long)B * 32;           k_blendT <<<gsz(n), 256, 0, stream>>>(blend, blT, B, n); }
    { long n = (long)D1 * 32;          k_biasB  <<<gsz(n), 256, 0, stream>>>(b1, b1B, D1,  D1,  n); }
    { long n = (long)D2 * 32;          k_biasB  <<<gsz(n), 256, 0, stream>>>(b2, b2B, D2,  D2,  n); }
    { long n = (long)D3P * 32;         k_biasB  <<<gsz(n), 256, 0, stream>>>(b3, b3B, D3,  D3P, n); }

    // layer 1: (B,480) -> (B,512), ELU, f16
    gemm_blend<0, 480, 512><<<dim3(B / 128, D1 / 128), 128, 0, stream>>>(
        x16, W1h, blT, b1B, blend, B, h1, D1);
    // layer 2: (B,512) -> (B,512), ELU, f16
    gemm_blend<0, 512, 512><<<dim3(B / 128, D2 / 128), 128, 0, stream>>>(
        h1, W2h, blT, b2B, blend, B, h2b, D2);
    // layer 3: (B,512) -> (B,384 padded), raw f32 logits
    gemm_blend<1, 512, 384><<<dim3(B / 128, D3P / 128), 128, 0, stream>>>(
        h2b, W3h, blT, b3B, blend, B, logits, D3P);

    // softmax over valid 363 columns, one wave per row
    softmax_rows<<<(B * 32 + 255) / 256, 256, 0, stream>>>(logits, out, B, D3, D3P);
}
